// GATModel_2345052144353
// MI455X (gfx1250) — matrix operand order repaired and checked by reading, not compiled, not run on hardware
//
#include <hip/hip_runtime.h>
#include <math.h>

// ---------------------------------------------------------------------------
// GAT (2-layer, DGL-style GATConv) for MI455X / gfx1250.
// Layer-1 GEMM (50000x128 @ 128x256) runs on v_wmma_f32_16x16x32_f16 with
// both operands pre-swizzled into WMMA fragment order so each lane loads its
// fragment as contiguous 32B (b128 pairs) instead of 16 scalar half-gathers.
// Edge softmax implemented as 3 segment-atomic passes (max / sum-exp / agg).
// ---------------------------------------------------------------------------

#define NN   50000
#define EE   800000
#define INF_ 128
#define HH   8
#define FF1  32
#define CC   2
#define HF1  (HH * FF1)      // 256
#define NEG_SLOPE 0.2f

typedef __attribute__((ext_vector_type(16))) _Float16 v16h;
typedef __attribute__((ext_vector_type(8)))  float    v8f;

static __device__ __forceinline__ void atomicMaxFloat(float* addr, float val) {
    // sign-aware int/uint trick: works for mixed signs, init must be -inf
    if (val >= 0.0f) {
        atomicMax((int*)addr, __float_as_int(val));
    } else {
        atomicMin((unsigned int*)addr, __float_as_uint(val));
    }
}

// ------------------------- small utility kernels ---------------------------

__global__ void fill_f32(float* p, int n, float v) {
    int i = blockIdx.x * blockDim.x + threadIdx.x;
    if (i < n) p[i] = v;
}

__global__ void init_out(float* out, const float* b2) {
    int i = blockIdx.x * blockDim.x + threadIdx.x;   // N*C
    if (i < NN * CC) out[i] = b2[i & (CC - 1)];
}

// W1[128,256] f32 -> f16, pre-swizzled into WMMA B-fragment order:
//   WhF[((kt*16 + nt)*32 + lane)*16 + e] = W1[k, n]
//   k = kt*32 + (lane>>4)*8 + (e&7) + (e>>3)*16   (B: lanes 0-15 hold K=0..7,16..23; 16-31 K=8..15,24..31)
//   n = nt*16 + (lane&15)          (lane holds column)
__global__ void convert_w1(const float* __restrict__ W1, _Float16* __restrict__ WhF) {
    int o = blockIdx.x * blockDim.x + threadIdx.x;   // 4*16*32*16 = 32768
    if (o >= INF_ * HF1) return;
    int e    = o & 15;
    int lane = (o >> 4) & 31;
    int nt   = (o >> 9) & 15;
    int kt   = o >> 13;
    int k = kt * 32 + (lane >> 4) * 8 + (e & 7) + ((e >> 3) << 4);
    int n = nt * 16 + (lane & 15);
    WhF[o] = (_Float16)W1[k * HF1 + n];
}

// ------------------------- layer 1 GEMM (WMMA) -----------------------------
// Z1[N,256] = X[N,128] @ W1[128,256].  Block: 256 thr = 8 waves, one 16-row
// M tile; each wave owns two 16-col N tiles (8 waves * 32 = 256 cols).
// A tile is staged in LDS already in per-lane fragment order so each lane
// reads its v16h fragment as 32 contiguous bytes.
__global__ void gemm1_wmma(const float* __restrict__ X,
                           const _Float16* __restrict__ WhF,
                           float* __restrict__ Z) {
    // AsF[kt][lane][e]: fragment-order A tile, 4*32*16 halves = 4 KB
    __shared__ __align__(32) _Float16 AsF[4][32][16];
    const int mtile = blockIdx.x;           // 0..3124 (50000/16 exact)
    const int t = threadIdx.x;

    // stage A tile (16 rows x 128 K) as f16 in fragment order.
    // (r, k) -> lane = r + 16*half, elem e; per ISA 16-bit A 16x32 layout:
    //   kl = k%32:  kl<8: half=0,e=kl | 8..15: half=1,e=kl-8
    //               16..23: half=0,e=kl-8 | 24..31: half=1,e=kl-16
    for (int i = t; i < 16 * INF_; i += 256) {
        int r = i >> 7, k = i & (INF_ - 1);
        int kt = k >> 5, kl = k & 31;
        int half = (kl >> 3) & 1;
        int e = (kl & 7) | ((kl >> 4) << 3);
        AsF[kt][r + half * 16][e] = (_Float16)X[(size_t)(mtile * 16 + r) * INF_ + k];
    }
    __syncthreads();

    const int wave = t >> 5;
    const int lane = t & 31;
    const int half_sel = lane >> 4;     // 0: lanes 0-15, 1: lanes 16-31
    const int l15 = lane & 15;          // column for B/C/D
    const int nt0 = wave * 2, nt1 = wave * 2 + 1;

    v8f acc0 = {}, acc1 = {};

    for (int kt = 0; kt < 4; ++kt) {    // K = 128 in chunks of 32
        v16h a  = *(const v16h*)&AsF[kt][lane][0];                     // ds b128 x2
        v16h b0 = *(const v16h*)(WhF + ((kt * 16 + nt0) * 32 + lane) * 16);
        v16h b1 = *(const v16h*)(WhF + ((kt * 16 + nt1) * 32 + lane) * 16);
        acc0 = __builtin_amdgcn_wmma_f32_16x16x32_f16(false, a, false, b0,
                                                      (short)0, acc0, false, false);
        acc1 = __builtin_amdgcn_wmma_f32_16x16x32_f16(false, a, false, b1,
                                                      (short)0, acc1, false, false);
    }

#pragma unroll
    for (int j = 0; j < 8; ++j) {       // C/D: VGPR j -> M=j (+8 for hi lanes)
        int row = mtile * 16 + j + half_sel * 8;
        Z[(size_t)row * HF1 + nt0 * 16 + l15] = acc0[j];
        Z[(size_t)row * HF1 + nt1 * 16 + l15] = acc1[j];
    }
}

// ------------------------- layer 1 attention coefficients ------------------

__global__ void attn1(const float* __restrict__ Z, const float* __restrict__ al,
                      const float* __restrict__ ar, float* el, float* er) {
    int tid = blockIdx.x * blockDim.x + threadIdx.x;   // N*H
    if (tid >= NN * HH) return;
    int n = tid >> 3, h = tid & 7;
    const float* zp = Z + (size_t)n * HF1 + h * FF1;
    const float* alp = al + h * FF1;
    const float* arp = ar + h * FF1;
    float sl = 0.f, sr = 0.f;
#pragma unroll 8
    for (int f = 0; f < FF1; ++f) {
        float z = zp[f];
        sl += z * alp[f];
        sr += z * arp[f];
    }
    el[tid] = sl;
    er[tid] = sr;
}

// ------------------------- layer 1 edge softmax ----------------------------

__global__ void edge_max1(const int* __restrict__ src, const int* __restrict__ dst,
                          const float* __restrict__ el, const float* __restrict__ er,
                          float* m) {
    int tid = blockIdx.x * blockDim.x + threadIdx.x;   // E*H
    if (tid >= EE * HH) return;
    int e = tid >> 3, h = tid & 7;
    int s = src[e], d = dst[e];
    float x = el[s * HH + h] + er[d * HH + h];
    x = x > 0.f ? x : NEG_SLOPE * x;
    atomicMaxFloat(&m[d * HH + h], x);
}

__global__ void edge_sum1(const int* __restrict__ src, const int* __restrict__ dst,
                          const float* __restrict__ el, const float* __restrict__ er,
                          const float* __restrict__ m, float* den, float* exbuf) {
    int tid = blockIdx.x * blockDim.x + threadIdx.x;   // E*H
    if (tid >= EE * HH) return;
    int e = tid >> 3, h = tid & 7;
    int s = src[e], d = dst[e];
    float x = el[s * HH + h] + er[d * HH + h];
    x = x > 0.f ? x : NEG_SLOPE * x;
    float ex = expf(x - m[d * HH + h]);
    atomicAdd(&den[d * HH + h], ex);
    exbuf[tid] = ex;
}

// one thread per (edge, 4-feature group): float4 load of z[src], 4 atomics
__global__ void edge_agg1(const int* __restrict__ src, const int* __restrict__ dst,
                          const float* __restrict__ Z, const float* __restrict__ den,
                          const float* __restrict__ exbuf, float* rst) {
    long long tid = (long long)blockIdx.x * blockDim.x + threadIdx.x;  // E*64
    if (tid >= (long long)EE * (HF1 / 4)) return;
    int e  = (int)(tid >> 6);
    int f4 = ((int)(tid & 63)) << 2;      // feature group start (16B aligned)
    int h  = f4 >> 5;
    int s = src[e], d = dst[e];
    float alpha = exbuf[e * HH + h] / den[d * HH + h];
    const float4 z = *(const float4*)(Z + (size_t)s * HF1 + f4);
    float* rp = rst + (size_t)d * HF1 + f4;
    atomicAdd(rp + 0, alpha * z.x);
    atomicAdd(rp + 1, alpha * z.y);
    atomicAdd(rp + 2, alpha * z.z);
    atomicAdd(rp + 3, alpha * z.w);
}

// ------------------------- ELU(rst + b1), in place -------------------------

__global__ void elu_bias(float* rst, const float* __restrict__ b1) {
    long long tid = (long long)blockIdx.x * blockDim.x + threadIdx.x;  // N*256
    if (tid >= (long long)NN * HF1) return;
    float x = rst[tid] + b1[tid & (HF1 - 1)];
    rst[tid] = x > 0.f ? x : (expf(x) - 1.f);
}

// ------------------------- layer 2 per-node (z2, el2, er2) -----------------

__global__ void layer2_node(const float* __restrict__ h1, const float* __restrict__ W2,
                            const float* __restrict__ al2, const float* __restrict__ ar2,
                            float* z2, float* el2, float* er2) {
    __shared__ float s0[256], s1[256];
    int n = blockIdx.x;              // one block per node
    int t = threadIdx.x;             // 256 threads = 256 features
    float v = h1[(size_t)n * HF1 + t];
    s0[t] = v * W2[t * CC + 0];
    s1[t] = v * W2[t * CC + 1];
    __syncthreads();
    for (int off = 128; off > 0; off >>= 1) {
        if (t < off) { s0[t] += s0[t + off]; s1[t] += s1[t + off]; }
        __syncthreads();
    }
    if (t == 0) {
        float z0 = s0[0], z1 = s1[0];
        z2[n * CC + 0] = z0;
        z2[n * CC + 1] = z1;
        el2[n] = z0 * al2[0] + z1 * al2[1];
        er2[n] = z0 * ar2[0] + z1 * ar2[1];
    }
}

// ------------------------- layer 2 edge softmax (H=1, C=2) -----------------

__global__ void edge_max2(const int* __restrict__ src, const int* __restrict__ dst,
                          const float* __restrict__ el2, const float* __restrict__ er2,
                          float* m2) {
    int e = blockIdx.x * blockDim.x + threadIdx.x;
    if (e >= EE) return;
    int s = src[e], d = dst[e];
    float x = el2[s] + er2[d];
    x = x > 0.f ? x : NEG_SLOPE * x;
    atomicMaxFloat(&m2[d], x);
}

__global__ void edge_sum2(const int* __restrict__ src, const int* __restrict__ dst,
                          const float* __restrict__ el2, const float* __restrict__ er2,
                          const float* __restrict__ m2, float* den2, float* ex2) {
    int e = blockIdx.x * blockDim.x + threadIdx.x;
    if (e >= EE) return;
    int s = src[e], d = dst[e];
    float x = el2[s] + er2[d];
    x = x > 0.f ? x : NEG_SLOPE * x;
    float ex = expf(x - m2[d]);
    atomicAdd(&den2[d], ex);
    ex2[e] = ex;
}

__global__ void edge_agg2(const int* __restrict__ src, const int* __restrict__ dst,
                          const float* __restrict__ z2, const float* __restrict__ den2,
                          const float* __restrict__ ex2, float* out) {
    int e = blockIdx.x * blockDim.x + threadIdx.x;
    if (e >= EE) return;
    int s = src[e], d = dst[e];
    float alpha = ex2[e] / den2[d];
    atomicAdd(&out[d * CC + 0], alpha * z2[s * CC + 0]);
    atomicAdd(&out[d * CC + 1], alpha * z2[s * CC + 1]);
}

// ---------------------------------------------------------------------------

static inline int cdiv(long long a, int b) { return (int)((a + b - 1) / b); }

extern "C" void kernel_launch(void* const* d_in, const int* in_sizes, int n_in,
                              void* d_out, int out_size, void* d_ws, size_t ws_size,
                              hipStream_t stream) {
    const float* x   = (const float*)d_in[0];
    const int*   src = (const int*)d_in[1];
    const int*   dst = (const int*)d_in[2];
    const float* W1  = (const float*)d_in[3];
    const float* al1 = (const float*)d_in[4];
    const float* ar1 = (const float*)d_in[5];
    const float* b1  = (const float*)d_in[6];
    const float* W2  = (const float*)d_in[7];
    const float* al2 = (const float*)d_in[8];
    const float* ar2 = (const float*)d_in[9];
    const float* b2  = (const float*)d_in[10];
    float* out = (float*)d_out;

    // ---- workspace layout (floats; every region is a multiple of 8 floats
    //      so the trailing f16 region stays 32B-aligned for v16h loads) ----
    float* f = (float*)d_ws;
    float* Z1    = f; f += (size_t)NN * HF1;   // 12.8M
    float* rst   = f; f += (size_t)NN * HF1;   // 12.8M (reused as h1 after ELU)
    float* el    = f; f += NN * HH;
    float* er    = f; f += NN * HH;
    float* m1    = f; f += NN * HH;
    float* den1  = f; f += NN * HH;
    float* exbuf = f; f += (size_t)EE * HH;    // 6.4M (reused as ex2)
    float* z2    = f; f += NN * CC;
    float* el2   = f; f += NN;
    float* er2   = f; f += NN;
    float* m2    = f; f += NN;
    float* den2  = f; f += NN;
    _Float16* WhF = (_Float16*)f;              // 128*256 halves, fragment order
    float* ex2 = exbuf;                        // layer-2 reuses exbuf

    const int B = 256;

    // --- layer 1 ---
    convert_w1<<<cdiv(INF_ * HF1, B), B, 0, stream>>>(W1, WhF);
    gemm1_wmma<<<NN / 16, B, 0, stream>>>(x, WhF, Z1);
    attn1<<<cdiv((long long)NN * HH, B), B, 0, stream>>>(Z1, al1, ar1, el, er);

    fill_f32<<<cdiv(NN * HH, B), B, 0, stream>>>(m1, NN * HH, -INFINITY);
    fill_f32<<<cdiv(NN * HH, B), B, 0, stream>>>(den1, NN * HH, 0.f);
    fill_f32<<<cdiv((long long)NN * HF1, B), B, 0, stream>>>(rst, NN * HF1, 0.f);

    edge_max1<<<cdiv((long long)EE * HH, B), B, 0, stream>>>(src, dst, el, er, m1);
    edge_sum1<<<cdiv((long long)EE * HH, B), B, 0, stream>>>(src, dst, el, er, m1, den1, exbuf);
    edge_agg1<<<cdiv((long long)EE * (HF1 / 4), B), B, 0, stream>>>(src, dst, Z1, den1, exbuf, rst);

    elu_bias<<<cdiv((long long)NN * HF1, B), B, 0, stream>>>(rst, b1);

    // --- layer 2 ---
    layer2_node<<<NN, B, 0, stream>>>(rst, W2, al2, ar2, z2, el2, er2);

    fill_f32<<<cdiv(NN, B), B, 0, stream>>>(m2, NN, -INFINITY);
    fill_f32<<<cdiv(NN, B), B, 0, stream>>>(den2, NN, 0.f);
    init_out<<<cdiv(NN * CC, B), B, 0, stream>>>(out, b2);  // out starts at b2

    edge_max2<<<cdiv(EE, B), B, 0, stream>>>(src, dst, el2, er2, m2);
    edge_sum2<<<cdiv(EE, B), B, 0, stream>>>(src, dst, el2, er2, m2, den2, ex2);
    edge_agg2<<<cdiv(EE, B), B, 0, stream>>>(src, dst, z2, den2, ex2, out);
}
